// Qwen3Experts_8358006358428
// MI455X (gfx1250) — compile-verified, matrix operand order checked by reading, not run on hardware
//
#include <hip/hip_runtime.h>
#include <math.h>

// ---------------- problem constants ----------------
constexpr int TOKENS  = 8192;
constexpr int HIDDEN  = 2048;
constexpr int INTER   = 768;
constexpr int NEXP    = 64;
constexpr int TOPK    = 8;
constexpr int TK      = TOKENS * TOPK;          // 65536 routed rows
constexpr int TILE_M  = 64;                     // rows per block tile (4 WMMA M-subtiles)
constexpr int MAX_TILES = TK / TILE_M + NEXP;   // 1088 (worst-case padding)
constexpr int PADROWS   = MAX_TILES * TILE_M;   // 69632

// ---------------- CDNA5 WMMA types ----------------
typedef __attribute__((ext_vector_type(16))) __bf16 bf16x16;
typedef __attribute__((ext_vector_type(8)))  float  f32x8;
typedef __attribute__((ext_vector_type(4)))  unsigned int u32x4;
typedef __attribute__((ext_vector_type(8)))  unsigned int u32x8;

__device__ __forceinline__ f32x8 wmma_bf16(bf16x16 a, bf16x16 b, f32x8 c) {
  // D = A(16x32 bf16) * B(32x16 bf16) + C(16x16 f32)
  return __builtin_amdgcn_wmma_f32_16x16x32_bf16(
      /*neg_a=*/false, a, /*neg_b=*/false, b,
      /*c_mod=*/(short)0, c, /*reuse_a=*/false, /*reuse_b=*/false);
}

__device__ __forceinline__ unsigned short f2bf(float f) {
  // round-to-nearest-even fp32 -> bf16 (manual)
  unsigned int u = __float_as_uint(f);
  u += 0x7FFFu + ((u >> 16) & 1u);
  return (unsigned short)(u >> 16);
}

#if defined(__has_builtin)
#if __has_builtin(__builtin_amdgcn_cvt_pk_bf16_f32)
#define HAVE_PK_BF16 1
#endif
#endif

__device__ __forceinline__ unsigned int f2bf2(float a, float b) {
  // pack two fp32 into one dword of bf16 (a -> low, b -> high)
#ifdef HAVE_PK_BF16
  typedef __attribute__((ext_vector_type(2))) __bf16 bf16x2;
  union { bf16x2 v; unsigned int u; } c;
  c.v = __builtin_amdgcn_cvt_pk_bf16_f32(a, b);
  return c.u;
#else
  return (unsigned int)f2bf(a) | ((unsigned int)f2bf(b) << 16);
#endif
}

__device__ __forceinline__ float bf2f(unsigned short s) {
  return __uint_as_float(((unsigned int)s) << 16);
}

// ---------------- 1. init ----------------
__global__ void moe_init_kernel(int* tile_e, int* row_map, int* counts, int* cursor) {
  int idx = blockIdx.x * blockDim.x + threadIdx.x;
  if (idx < MAX_TILES) tile_e[idx] = -1;
  if (idx < PADROWS)   row_map[idx] = -1;
  if (idx < NEXP) { counts[idx] = 0; cursor[idx] = 0; }
}

// ---------------- 2. routing: top-8 + softmax (one wave32 per token) ----------------
__global__ __launch_bounds__(256) void moe_route_kernel(
    const float* __restrict__ logits, int* __restrict__ sel,
    float* __restrict__ wts, int* __restrict__ counts) {
  int wave = threadIdx.x >> 5;
  int lane = threadIdx.x & 31;
  int t = blockIdx.x * 8 + wave;
  const float* lg = logits + (size_t)t * NEXP;
  float v0 = lg[lane];
  float v1 = lg[lane + 32];

  float topv[TOPK];
  int   topi[TOPK];
#pragma unroll
  for (int it = 0; it < TOPK; ++it) {
    float bv; int bi;
    if (v0 >= v1) { bv = v0; bi = lane; } else { bv = v1; bi = lane + 32; }
#pragma unroll
    for (int off = 16; off >= 1; off >>= 1) {
      float ov = __shfl_xor(bv, off, 32);
      int   oi = __shfl_xor(bi, off, 32);
      if (ov > bv || (ov == bv && oi < bi)) { bv = ov; bi = oi; }
    }
    topv[it] = bv; topi[it] = bi;
    if (bi == lane)      v0 = -INFINITY;
    if (bi == lane + 32) v1 = -INFINITY;
  }
  float m = topv[0];
  float s = 0.f;
#pragma unroll
  for (int it = 0; it < TOPK; ++it) { topv[it] = __expf(topv[it] - m); s += topv[it]; }
  float inv = 1.f / s;
  if (lane == 0) {
#pragma unroll
    for (int it = 0; it < TOPK; ++it) {
      sel[t * TOPK + it] = topi[it];
      wts[t * TOPK + it] = topv[it] * inv;
      atomicAdd(&counts[topi[it]], 1);
    }
  }
}

// ---------------- 3. serial scan: padded group offsets + tile->expert map ----------------
__global__ void moe_scan_kernel(const int* __restrict__ counts,
                                int* __restrict__ eoff, int* __restrict__ tile_e) {
  if (threadIdx.x == 0 && blockIdx.x == 0) {
    int off = 0;
    for (int e = 0; e < NEXP; ++e) {
      eoff[e] = off;
      int tiles = (counts[e] + TILE_M - 1) / TILE_M;
      int t0 = off / TILE_M;
      for (int t = 0; t < tiles; ++t) tile_e[t0 + t] = e;
      off += tiles * TILE_M;
    }
  }
}

// ---------------- 4. scatter assignments into expert-grouped (padded) row space ----------------
__global__ void moe_scatter_kernel(const int* __restrict__ sel,
                                   const int* __restrict__ eoff,
                                   int* __restrict__ cursor,
                                   int* __restrict__ row_map) {
  int i = blockIdx.x * blockDim.x + threadIdx.x;
  if (i >= TK) return;
  int e = sel[i];
  int p = eoff[e] + atomicAdd(&cursor[e], 1);
  row_map[p] = i;  // i = token*TOPK + k
}

// ---------------- 5. grouped GEMM: gate & up + fused SwiGLU -> bf16 h ----------------
// block = 256 thr (8 waves). grid = (MAX_TILES, INTER/128).
// Block tile: 64 rows x 128 cols. Each wave: 16 cols, 4 M-subtiles reusing one B fragment
// (8 WMMAs per B-pair -> 4x less B traffic than 16-row tiles).
// A-tile row-gather by the Tensor Data Mover in gather mode (4 descriptors x 16 indices).
__global__ __launch_bounds__(256) void moe_gateup_kernel(
    const float* __restrict__ x, const float* __restrict__ Wg,
    const float* __restrict__ Wu, const int* __restrict__ tile_e,
    const int* __restrict__ row_map, unsigned short* __restrict__ hbuf) {
  __shared__ int rowtok[TILE_M];
  __shared__ float Astage[TILE_M][64];       // TDM gather dest: 64 rows x 64 fp32 (16KB)
  __shared__ unsigned short As[TILE_M][80];  // bf16 fragment source, padded stride (160B rows)

  int tile = blockIdx.x;
  int e = tile_e[tile];
  if (e < 0) return;
  int row0 = tile * TILE_M;
  int tid = threadIdx.x;
  if (tid < TILE_M) {
    int i = row_map[row0 + tid];
    rowtok[tid] = (i >= 0) ? (i >> 3) : 0;  // pad rows: gather token 0, masked downstream
  }
  __syncthreads();

  int wave = tid >> 5, lane = tid & 31;
  int hi = lane >> 4, ln = lane & 15;
  int colBase = blockIdx.y * 128 + wave * 16;
  int col = colBase + ln;
  const size_t wstride = (size_t)HIDDEN * INTER;
  const float* Bg = Wg + (size_t)e * wstride + col;
  const float* Bu = Wu + (size_t)e * wstride + col;

  // ---- Tensor DMA descriptors (wave 0 only; all fields wave-uniform) ----
  // Group1: data_size=4B(code2); tensor_dim0=HIDDEN; tensor_dim1=TOKENS;
  //         tile_dim0=64; tile_dim1=16 gather indices; dim0_stride=HIDDEN.
  u32x8 g1v;
  u32x4 g2q[4], g3q[4];
  unsigned int ldsOff = 0;
  if (wave == 0) {
    g1v[0] = 0x00020000u;                          // data_size=2 (4 bytes) at [17:16]
    g1v[1] = ((unsigned)HIDDEN & 0xFFFFu) << 16;   // tensor_dim0[15:0] at bits 63:48
    g1v[2] = ((unsigned)TOKENS & 0xFFFFu) << 16;   // tensor_dim1[15:0] at bits 95:80
    g1v[3] = 64u << 16;                            // tile_dim0=64 at bits 127:112
    g1v[4] = 16u;                                  // tile_dim1=16 (#valid indices)
    g1v[5] = (unsigned)HIDDEN;                     // tensor_dim0_stride low32
    g1v[6] = 0u;
    g1v[7] = 0u;
    // Four quarters of 16 token indices each, packed two 16-bit indices per dword.
#pragma unroll
    for (int q = 0; q < 4; ++q) {
      unsigned int pr = 0;
      if (lane < 8) {
        unsigned a = (unsigned)rowtok[q * 16 + 2 * lane] & 0xFFFFu;
        unsigned b = (unsigned)rowtok[q * 16 + 2 * lane + 1] & 0xFFFFu;
        pr = a | (b << 16);
      }
      g2q[q][0] = (unsigned)__builtin_amdgcn_readlane((int)pr, 0);
      g2q[q][1] = (unsigned)__builtin_amdgcn_readlane((int)pr, 1);
      g2q[q][2] = (unsigned)__builtin_amdgcn_readlane((int)pr, 2);
      g2q[q][3] = (unsigned)__builtin_amdgcn_readlane((int)pr, 3);
      g3q[q][0] = (unsigned)__builtin_amdgcn_readlane((int)pr, 4);
      g3q[q][1] = (unsigned)__builtin_amdgcn_readlane((int)pr, 5);
      g3q[q][2] = (unsigned)__builtin_amdgcn_readlane((int)pr, 6);
      g3q[q][3] = (unsigned)__builtin_amdgcn_readlane((int)pr, 7);
    }
    ldsOff = (unsigned int)(size_t)&Astage[0][0];  // LDS byte offset of staging tile
  }
  unsigned long long xbase = (unsigned long long)(size_t)x;

  f32x8 accg[4] = {};
  f32x8 accu[4] = {};

  for (int kb = 0; kb < HIDDEN; kb += 64) {
    // ---- TDM gather: 4 x (16 token rows x 64 fp32) -> Astage ----
    if (wave == 0) {
      unsigned long long ga = xbase + (unsigned long long)kb * 4ull;  // tile column offset
#pragma unroll
      for (int q = 0; q < 4; ++q) {
        u32x4 g0v;
        g0v[0] = 1u | (1u << 31);                        // count=1, gather_mode=1, 16-bit idx
        g0v[1] = ldsOff + 4096u * q;                     // lds_addr (+16 rows x 256B per quarter)
        g0v[2] = (unsigned)(ga & 0xFFFFFFFFull);         // global_addr[31:0]
        g0v[3] = (unsigned)((ga >> 32) & 0x01FFFFFFull)  // global_addr[56:32]
                 | (2u << 30);                           // type=2 ("image")
        asm volatile("tensor_load_to_lds %0, %1, %2, %3"
                     :: "s"(g0v), "s"(g1v), "s"(g2q[q]), "s"(g3q[q])
                     : "memory");
      }
      __builtin_amdgcn_s_wait_tensorcnt(0);
    }
    __syncthreads();

    // ---- convert staged fp32 -> bf16 fragment layout (each thread: one row's 16 elems) ----
    {
      int idx16 = tid * 16;
      int m = idx16 >> 6;
      int c = idx16 & 63;
#pragma unroll
      for (int p = 0; p < 4; ++p) {
        float4 v = *reinterpret_cast<const float4*>(&Astage[m][c + 4 * p]);
        *reinterpret_cast<unsigned int*>(&As[m][c + 4 * p])     = f2bf2(v.x, v.y);
        *reinterpret_cast<unsigned int*>(&As[m][c + 4 * p + 2]) = f2bf2(v.z, v.w);
      }
    }
    __syncthreads();

    if (kb + 64 < HIDDEN) {  // hint next weight slab toward L2
      __builtin_prefetch(Bg + (size_t)(kb + 64) * INTER, 0, 1);
      __builtin_prefetch(Bu + (size_t)(kb + 64) * INTER, 0, 1);
    }

#pragma unroll
    for (int s = 0; s < 2; ++s) {
      int ko = s * 32;
      int kbase = kb + ko;
      // B fragments: lane=N, gather 16 strided fp32 rows, packed bf16 convert
      union { unsigned int u32[8]; bf16x16 v; } bg, bu;
#pragma unroll
      for (int j = 0; j < 4; ++j) {
        bg.u32[j]     = f2bf2(Bg[(size_t)(kbase + hi * 8 + 2 * j) * INTER],
                              Bg[(size_t)(kbase + hi * 8 + 2 * j + 1) * INTER]);
        bg.u32[4 + j] = f2bf2(Bg[(size_t)(kbase + 16 + hi * 8 + 2 * j) * INTER],
                              Bg[(size_t)(kbase + 16 + hi * 8 + 2 * j + 1) * INTER]);
        bu.u32[j]     = f2bf2(Bu[(size_t)(kbase + hi * 8 + 2 * j) * INTER],
                              Bu[(size_t)(kbase + hi * 8 + 2 * j + 1) * INTER]);
        bu.u32[4 + j] = f2bf2(Bu[(size_t)(kbase + 16 + hi * 8 + 2 * j) * INTER],
                              Bu[(size_t)(kbase + 16 + hi * 8 + 2 * j + 1) * INTER]);
      }
      // 4 M-subtiles reuse the same B fragments: 8 WMMAs per B-pair
#pragma unroll
      for (int m = 0; m < 4; ++m) {
        union { uint4 u4[2]; bf16x16 v; } af;
        af.u4[0] = *reinterpret_cast<const uint4*>(&As[m * 16 + ln][ko + hi * 8]);
        af.u4[1] = *reinterpret_cast<const uint4*>(&As[m * 16 + ln][ko + 16 + hi * 8]);
        accg[m] = wmma_bf16(af.v, bg.v, accg[m]);
        accu[m] = wmma_bf16(af.v, bu.v, accu[m]);
      }
    }
    __syncthreads();  // all fragment reads done before next TDM overwrite
  }

  // fused SwiGLU, store bf16 h. C/D layout: lane=N(=ln), M = hi*8 + vgpr
#pragma unroll
  for (int m = 0; m < 4; ++m) {
#pragma unroll
    for (int j = 0; j < 8; ++j) {
      int M = m * 16 + hi * 8 + j;
      float g = accg[m][j], u = accu[m][j];
      float hval = (g / (1.f + __expf(-g))) * u;
      hbuf[(size_t)(row0 + M) * INTER + col] = f2bf(hval);
    }
  }
}

// ---------------- 6. grouped GEMM: down; scatter rows to unsorted per-assignment buffer ----------------
// Block tile: 64 rows x 128 cols; each wave 16 cols x 4 M-subtiles sharing one B fragment.
__global__ __launch_bounds__(256) void moe_down_kernel(
    const unsigned short* __restrict__ hbuf, const float* __restrict__ Wd,
    const int* __restrict__ tile_e, const int* __restrict__ row_map,
    unsigned short* __restrict__ dbuf) {
  __shared__ int rowi[TILE_M];
  int tile = blockIdx.x;
  int e = tile_e[tile];
  if (e < 0) return;
  int row0 = tile * TILE_M;
  int tid = threadIdx.x;
  if (tid < TILE_M) rowi[tid] = row_map[row0 + tid];
  __syncthreads();

  int wave = tid >> 5, lane = tid & 31;
  int hi = lane >> 4, ln = lane & 15;
  int colBase = blockIdx.y * 128 + wave * 16;
  int col = colBase + ln;
  const float* Bd = Wd + (size_t)e * INTER * HIDDEN + col;

  f32x8 acc[4] = {};
  for (int kb = 0; kb < INTER; kb += 32) {
    union { unsigned int u32[8]; bf16x16 v; } bd;
#pragma unroll
    for (int j = 0; j < 4; ++j) {
      bd.u32[j]     = f2bf2(Bd[(size_t)(kb + hi * 8 + 2 * j) * HIDDEN],
                            Bd[(size_t)(kb + hi * 8 + 2 * j + 1) * HIDDEN]);
      bd.u32[4 + j] = f2bf2(Bd[(size_t)(kb + 16 + hi * 8 + 2 * j) * HIDDEN],
                            Bd[(size_t)(kb + 16 + hi * 8 + 2 * j + 1) * HIDDEN]);
    }
#pragma unroll
    for (int m = 0; m < 4; ++m) {
      const unsigned short* hrow = hbuf + (size_t)(row0 + m * 16 + ln) * INTER;
      union { uint4 u4[2]; bf16x16 v; } af;  // A already bf16; b128 loads match WMMA layout
      af.u4[0] = *reinterpret_cast<const uint4*>(hrow + kb + hi * 8);
      af.u4[1] = *reinterpret_cast<const uint4*>(hrow + kb + 16 + hi * 8);
      acc[m] = wmma_bf16(af.v, bd.v, acc[m]);
    }
  }

#pragma unroll
  for (int m = 0; m < 4; ++m) {
#pragma unroll
    for (int j = 0; j < 8; ++j) {
      int M = m * 16 + hi * 8 + j;
      int i = rowi[M];
      if (i >= 0) dbuf[(size_t)i * HIDDEN + col] = f2bf(acc[m][j]);
    }
  }
}

// ---------------- 7. deterministic weighted combine over top-k ----------------
__global__ __launch_bounds__(256) void moe_combine_kernel(
    const unsigned short* __restrict__ dbuf, const float* __restrict__ wts,
    float* __restrict__ out) {
  int idx = blockIdx.x * blockDim.x + threadIdx.x;  // token*HIDDEN + col
  if (idx >= TOKENS * HIDDEN) return;
  int t = idx >> 11;        // / HIDDEN
  int c = idx & (HIDDEN - 1);
  float s = 0.f;
#pragma unroll
  for (int k = 0; k < TOPK; ++k) {
    int i = t * TOPK + k;
    s += wts[i] * bf2f(dbuf[(size_t)i * HIDDEN + c]);
  }
  out[idx] = s;
}

// ---------------- host launcher ----------------
extern "C" void kernel_launch(void* const* d_in, const int* in_sizes, int n_in,
                              void* d_out, int out_size, void* d_ws, size_t ws_size,
                              hipStream_t stream) {
  (void)in_sizes; (void)n_in; (void)out_size; (void)ws_size;
  const float* x  = (const float*)d_in[0];   // hidden_states [T,H]
  const float* rl = (const float*)d_in[1];   // router_logits [T,E]
  const float* Wg = (const float*)d_in[2];   // gate_proj [E,H,I]
  const float* Wu = (const float*)d_in[3];   // up_proj   [E,H,I]
  const float* Wd = (const float*)d_in[4];   // down_proj [E,I,H]
  float* out = (float*)d_out;                // [T,H]

  char* ws = (char*)d_ws;
  size_t off = 0;
  auto alloc = [&](size_t bytes) -> char* {
    char* p = ws + off;
    off = (off + bytes + 255) & ~(size_t)255;
    return p;
  };
  int*   sel     = (int*)alloc((size_t)TK * 4);
  float* wts     = (float*)alloc((size_t)TK * 4);
  int*   counts  = (int*)alloc(NEXP * 4);
  int*   cursor  = (int*)alloc(NEXP * 4);
  int*   eoff    = (int*)alloc(NEXP * 4);
  int*   tile_e  = (int*)alloc((size_t)MAX_TILES * 4);
  int*   row_map = (int*)alloc((size_t)PADROWS * 4);
  unsigned short* hbuf = (unsigned short*)alloc((size_t)PADROWS * INTER * 2);   // ~107 MB
  unsigned short* dbuf = (unsigned short*)alloc((size_t)TK * HIDDEN * 2);       // 256 MB

  moe_init_kernel<<<(PADROWS + 255) / 256, 256, 0, stream>>>(tile_e, row_map, counts, cursor);
  moe_route_kernel<<<TOKENS / 8, 256, 0, stream>>>(rl, sel, wts, counts);
  moe_scan_kernel<<<1, 32, 0, stream>>>(counts, eoff, tile_e);
  moe_scatter_kernel<<<TK / 256, 256, 0, stream>>>(sel, eoff, cursor, row_map);

  dim3 g1(MAX_TILES, INTER / 128);
  moe_gateup_kernel<<<g1, 256, 0, stream>>>(x, Wg, Wu, tile_e, row_map, hbuf);

  dim3 g2(MAX_TILES, HIDDEN / 128);
  moe_down_kernel<<<g2, 256, 0, stream>>>(hbuf, Wd, tile_e, row_map, dbuf);

  moe_combine_kernel<<<(TOKENS * HIDDEN + 255) / 256, 256, 0, stream>>>(dbuf, wts, out);
}
